// GATRec_75539884802518
// MI455X (gfx1250) — matrix operand order repaired and checked by reading, not compiled, not run on hardware
//
#include <hip/hip_runtime.h>
#include <hip/hip_bf16.h>

// ---------- types for CDNA5 WMMA ----------
typedef __attribute__((ext_vector_type(16))) __bf16          v16bf;
typedef __attribute__((ext_vector_type(16))) unsigned short  v16us;
typedef __attribute__((ext_vector_type(8)))  float           v8f;

#define NEG_SLOPE 0.2f

// float -> bf16 round-to-nearest-even
static __device__ __forceinline__ unsigned short f2bf(float f) {
  unsigned int u = __float_as_uint(f);
  u += 0x7FFFu + ((u >> 16) & 1u);
  return (unsigned short)(u >> 16);
}
// order-preserving float<->int mapping for atomic float max via atomicMax(int)
static __device__ __forceinline__ int ordf(float f) {
  int i = __float_as_int(f);
  return (i < 0) ? (i ^ 0x7FFFFFFF) : i;
}
static __device__ __forceinline__ float unordf(int i) {
  return __int_as_float((i < 0) ? (i ^ 0x7FFFFFFF) : i);
}

// ---------- fill kernels ----------
__global__ void k_fill_f32(float* p, float v, size_t n) {
  size_t i = blockIdx.x * (size_t)blockDim.x + threadIdx.x;
  size_t st = gridDim.x * (size_t)blockDim.x;
  for (; i < n; i += st) p[i] = v;
}
__global__ void k_fill_i32(int* p, int v, size_t n) {
  size_t i = blockIdx.x * (size_t)blockDim.x + threadIdx.x;
  size_t st = gridDim.x * (size_t)blockDim.x;
  for (; i < n; i += st) p[i] = v;
}

// ---------- build bf16 node features (concat user/item) ----------
__global__ void k_build_xb(const float* xu, const float* xi, unsigned short* xb,
                           int NU, size_t n) {  // n = N*64
  size_t i = blockIdx.x * (size_t)blockDim.x + threadIdx.x;
  size_t st = gridDim.x * (size_t)blockDim.x;
  for (; i < n; i += st) {
    size_t row = i >> 6; int c = (int)(i & 63);
    float v = ((int)row < NU) ? xu[row * 64 + c] : xi[(row - NU) * 64 + c];
    xb[i] = f2bf(v);
  }
}

// ---------- weight convert+transpose: Wt[n*K + k] = bf16(W[k*Nc + n]) ----------
__global__ void k_cvt_w(const float* W, unsigned short* Wt, int K, int Nc) {
  int t = blockIdx.x * blockDim.x + threadIdx.x;
  if (t >= K * Nc) return;
  int n = t / K, k = t % K;
  Wt[t] = f2bf(W[(size_t)k * Nc + n]);
}

// ---------- WMMA bf16 GEMM: C[M x Nc] = A[M x K] * Bt^T (Bt is [Nc x K]) ----------
// blockDim.x == 256 (8 waves). Each wave computes a 16x64 strip: one A fragment
// per K-chunk feeds 4 independent back-to-back WMMAs (4 accumulators), so the
// XDL pipe stays busy and A is fetched once per K-chunk instead of once per
// 16-col tile. B panel (weights) is tiny and stays cache-resident.
__global__ void k_wmma_gemm_bf16(const unsigned short* __restrict__ A,
                                 const unsigned short* __restrict__ Bt,
                                 float* __restrict__ Cm, int M, int Nc, int K) {
  const int lane = threadIdx.x & 31;
  const int wave = threadIdx.x >> 5;
  const int mr = lane & 15, half = lane >> 4;
  const int tm = blockIdx.x * 8 + wave;          // 16-row tile index
  const int tn0 = blockIdx.y * 4;                // first 16-col tile of this strip
  if (tm * 16 >= M) return;                      // whole wave uniform -> safe

  int arow = tm * 16 + mr; if (arow >= M) arow = M - 1;   // clamp tail rows
  const unsigned short* ap = A + (size_t)arow * K;
  const unsigned short* bp0 = Bt + (size_t)(tn0 * 16 + mr) * K;

  v8f acc0 = {}, acc1 = {}, acc2 = {}, acc3 = {};
  for (int kc = 0; kc < K; kc += 32) {
    v16us au, bu0, bu1, bu2, bu3;
#pragma unroll
    for (int v = 0; v < 8; ++v) {
      // A 16x32 bf16 fragment: lanes 0-15 hold K=0..15 (2/VGPR), lanes 16-31 hold K=16..31
      int ka = kc + 16 * half + 2 * v;
      au[2 * v]     = ap[ka];
      au[2 * v + 1] = ap[ka + 1];
      // B 32x16 fragment: lanes 0-15 hold K=0..15 (2/VGPR), lanes 16-31 hold K=16..31
      int kb = kc + 16 * half + 2 * v;
      bu0[2 * v] = bp0[kb];           bu0[2 * v + 1] = bp0[kb + 1];
      bu1[2 * v] = bp0[kb + 16 * K];  bu1[2 * v + 1] = bp0[kb + 16 * K + 1];
      bu2[2 * v] = bp0[kb + 32 * K];  bu2[2 * v + 1] = bp0[kb + 32 * K + 1];
      bu3[2 * v] = bp0[kb + 48 * K];  bu3[2 * v + 1] = bp0[kb + 48 * K + 1];
    }
    v16bf a  = __builtin_bit_cast(v16bf, au);
    v16bf b0 = __builtin_bit_cast(v16bf, bu0);
    v16bf b1 = __builtin_bit_cast(v16bf, bu1);
    v16bf b2 = __builtin_bit_cast(v16bf, bu2);
    v16bf b3 = __builtin_bit_cast(v16bf, bu3);
    // 4 independent WMMAs (distinct accumulators): no D->A/B hazard, back-to-back issue
    acc0 = __builtin_amdgcn_wmma_f32_16x16x32_bf16(false, a, false, b0, (short)0, acc0, false, false);
    acc1 = __builtin_amdgcn_wmma_f32_16x16x32_bf16(false, a, false, b1, (short)0, acc1, false, false);
    acc2 = __builtin_amdgcn_wmma_f32_16x16x32_bf16(false, a, false, b2, (short)0, acc2, false, false);
    acc3 = __builtin_amdgcn_wmma_f32_16x16x32_bf16(false, a, false, b3, (short)0, acc3, false, false);
  }
#pragma unroll
  for (int r = 0; r < 8; ++r) {
    int row = tm * 16 + r + 8 * half;  // C/D layout: VGPR r -> M=r (lanes<16) / r+8 (lanes>=16)
    if (row < M) {
      float* crow = Cm + (size_t)row * Nc + tn0 * 16 + mr;
      crow[0]  = acc0[r];
      crow[16] = acc1[r];
      crow[32] = acc2[r];
      crow[48] = acc3[r];
    }
  }
}

// ---------- per-node attention dots: a[n,h] = sum_c h[n,h,c]*att[h,c]  (C==64) ----------
__global__ void k_att(const float* __restrict__ hm, const float* __restrict__ attS,
                      const float* __restrict__ attD, float* aS, float* aD, int N, int H) {
  int w = (int)((blockIdx.x * (size_t)blockDim.x + threadIdx.x) >> 5);  // one wave per (node, head)
  int lane = threadIdx.x & 31;
  int node = w / H, h = w % H;
  if (node >= N) return;
  const float* hp = hm + (size_t)node * H * 64 + (size_t)h * 64;
  float s0 = hp[lane] * attS[h * 64 + lane] + hp[lane + 32] * attS[h * 64 + lane + 32];
  float s1 = hp[lane] * attD[h * 64 + lane] + hp[lane + 32] * attD[h * 64 + lane + 32];
#pragma unroll
  for (int off = 16; off; off >>= 1) {
    s0 += __shfl_xor(s0, off, 32);
    s1 += __shfl_xor(s1, off, 32);
  }
  if (lane == 0) { aS[node * H + h] = s0; aD[node * H + h] = s1; }
}

// ---------- edge pass 1: segment max of leaky_relu logits (ordered-int atomicMax) ----------
__global__ void k_edge_max(const int* __restrict__ src, const int* __restrict__ dst,
                           int E, int Etot, int H,
                           const float* __restrict__ as, const float* __restrict__ ad, int* m) {
  int t = blockIdx.x * blockDim.x + threadIdx.x;
  if (t >= Etot * H) return;
  int e = t / H, h = t - e * H;
  int s, d;
  if (e < E) { s = src[e]; d = dst[e]; } else { s = d = e - E; }  // self-loops
  float l = as[s * H + h] + ad[d * H + h];
  l = l > 0.f ? l : NEG_SLOPE * l;
  atomicMax(&m[d * H + h], ordf(l));
}

// ---------- edge pass 2: segment sum of exp(logit - max) ----------
__global__ void k_edge_sum(const int* __restrict__ src, const int* __restrict__ dst,
                           int E, int Etot, int H,
                           const float* __restrict__ as, const float* __restrict__ ad,
                           const int* __restrict__ m, float* ssum) {
  int t = blockIdx.x * blockDim.x + threadIdx.x;
  if (t >= Etot * H) return;
  int e = t / H, h = t - e * H;
  int s, d;
  if (e < E) { s = src[e]; d = dst[e]; } else { s = d = e - E; }
  float l = as[s * H + h] + ad[d * H + h];
  l = l > 0.f ? l : NEG_SLOPE * l;
  atomicAdd(&ssum[d * H + h], __expf(l - unordf(m[d * H + h])));
}

// ---------- edge pass 3 (layer1, H=4,C=64): scatter h[src]*alpha; one block per edge ----------
__global__ void k_edge_scat4(const int* __restrict__ src, const int* __restrict__ dst, int E,
                             const float* __restrict__ as, const float* __restrict__ ad,
                             const int* __restrict__ m, const float* __restrict__ ssum,
                             const float* __restrict__ hsrc, float* __restrict__ outp) {
  int e = blockIdx.x;
  int s, d;
  if (e < E) { s = src[e]; d = dst[e]; } else { s = d = e - E; }
  int idx = threadIdx.x;        // 0..255 -> h = idx>>6, c = idx&63
  int h = idx >> 6;
  float l = as[s * 4 + h] + ad[d * 4 + h];
  l = l > 0.f ? l : NEG_SLOPE * l;
  float alpha = __expf(l - unordf(m[d * 4 + h])) / (ssum[d * 4 + h] + 1e-16f);
  atomicAdd(&outp[(size_t)d * 256 + idx], hsrc[(size_t)s * 256 + idx] * alpha);
}

// ---------- edge pass 3 (layer2, H=1,C=64): 4 edges per 256-thread block ----------
__global__ void k_edge_scat1(const int* __restrict__ src, const int* __restrict__ dst,
                             int E, int Etot,
                             const float* __restrict__ as, const float* __restrict__ ad,
                             const int* __restrict__ m, const float* __restrict__ ssum,
                             const float* __restrict__ hsrc, float* __restrict__ outp) {
  int e = blockIdx.x * 4 + (threadIdx.x >> 6);
  if (e >= Etot) return;
  int c = threadIdx.x & 63;
  int s, d;
  if (e < E) { s = src[e]; d = dst[e]; } else { s = d = e - E; }
  float l = as[s] + ad[d];
  l = l > 0.f ? l : NEG_SLOPE * l;
  float alpha = __expf(l - unordf(m[d])) / (ssum[d] + 1e-16f);
  atomicAdd(&outp[(size_t)d * 64 + c], hsrc[(size_t)s * 64 + c] * alpha);
}

// ---------- bias + ELU + bf16 convert for layer-2 input ----------
__global__ void k_post1(const float* __restrict__ outp, const float* __restrict__ b,
                        unsigned short* x2b, size_t n) {  // n = N*256
  size_t i = blockIdx.x * (size_t)blockDim.x + threadIdx.x;
  size_t st = gridDim.x * (size_t)blockDim.x;
  for (; i < n; i += st) {
    float v = outp[i] + b[i & 255];
    v = v > 0.f ? v : (__expf(v) - 1.f);
    x2b[i] = f2bf(v);
  }
}

// ---------- final bias add into d_out ----------
__global__ void k_post2(float* __restrict__ o, const float* __restrict__ b, size_t n) { // n = N*64
  size_t i = blockIdx.x * (size_t)blockDim.x + threadIdx.x;
  size_t st = gridDim.x * (size_t)blockDim.x;
  for (; i < n; i += st) o[i] += b[i & 63];
}

extern "C" void kernel_launch(void* const* d_in, const int* in_sizes, int n_in,
                              void* d_out, int out_size, void* d_ws, size_t ws_size,
                              hipStream_t stream) {
  const int E  = in_sizes[0] / 2;
  const int NU = in_sizes[1] / 64;
  const int NI = in_sizes[2] / 64;
  const int N  = NU + NI;                 // 110000 = 6875*16 (multiple of 16)
  const int IN = 64, H = 4, F1 = 256, OUT = 64;
  const int Etot = E + N;

  const int*   ei   = (const int*)d_in[0];
  const int*   src  = ei;
  const int*   dst  = ei + E;
  const float* xu   = (const float*)d_in[1];
  const float* xi   = (const float*)d_in[2];
  const float* W1   = (const float*)d_in[3];
  const float* at_s1= (const float*)d_in[4];
  const float* at_d1= (const float*)d_in[5];
  const float* b1   = (const float*)d_in[6];
  const float* W2   = (const float*)d_in[7];
  const float* at_s2= (const float*)d_in[8];
  const float* at_d2= (const float*)d_in[9];
  const float* b2   = (const float*)d_in[10];
  float* out = (float*)d_out;

  // ---- workspace carve (~333 MB) ----
  char* ws = (char*)d_ws;
  size_t off = 0;
  auto carve = [&](size_t bytes) -> void* {
    void* p = ws + off;
    off = (off + bytes + 255) & ~(size_t)255;
    return p;
  };
  unsigned short* xb  = (unsigned short*)carve((size_t)N * IN * 2);
  unsigned short* w1b = (unsigned short*)carve((size_t)F1 * IN * 2);   // [256][64]
  unsigned short* w2b = (unsigned short*)carve((size_t)OUT * F1 * 2);  // [64][256]
  float* h1   = (float*)carve((size_t)N * F1 * 4);
  float* out1 = (float*)carve((size_t)N * F1 * 4);
  float* as1  = (float*)carve((size_t)N * H * 4);
  float* ad1  = (float*)carve((size_t)N * H * 4);
  int*   m1   = (int*)  carve((size_t)N * H * 4);
  float* s1   = (float*)carve((size_t)N * H * 4);
  unsigned short* x2b = (unsigned short*)carve((size_t)N * F1 * 2);
  float* h2   = (float*)carve((size_t)N * OUT * 4);
  float* as2  = (float*)carve((size_t)N * 4);
  float* ad2  = (float*)carve((size_t)N * 4);
  int*   m2   = (int*)  carve((size_t)N * 4);
  float* s2   = (float*)carve((size_t)N * 4);
  (void)ws_size; (void)n_in; (void)out_size;

  const int TB = 256;
  const int FILLB = 2048;

  // ---- inputs -> bf16 ----
  k_build_xb<<<FILLB, TB, 0, stream>>>(xu, xi, xb, NU, (size_t)N * IN);
  k_cvt_w<<<(IN * F1 + TB - 1) / TB, TB, 0, stream>>>(W1, w1b, IN, F1);
  k_cvt_w<<<(F1 * OUT + TB - 1) / TB, TB, 0, stream>>>(W2, w2b, F1, OUT);

  // ---- GEMM1: h1 = x @ W1 (bf16 WMMA, f32 accumulate); waves do 16x64 strips ----
  {
    int mtiles = (N + 15) / 16;
    k_wmma_gemm_bf16<<<dim3((mtiles + 7) / 8, F1 / 64), TB, 0, stream>>>(xb, w1b, h1, N, F1, IN);
  }

  // ---- layer-1 attention coefficients ----
  k_att<<<((size_t)N * H + 7) / 8, TB, 0, stream>>>(h1, at_s1, at_d1, as1, ad1, N, H);

  // ---- layer-1 segment softmax + scatter ----
  k_fill_i32<<<FILLB, TB, 0, stream>>>(m1, (int)0x80000000, (size_t)N * H);
  k_fill_f32<<<FILLB, TB, 0, stream>>>(s1, 0.f, (size_t)N * H);
  k_fill_f32<<<FILLB, TB, 0, stream>>>(out1, 0.f, (size_t)N * F1);
  k_edge_max<<<((size_t)Etot * H + TB - 1) / TB, TB, 0, stream>>>(src, dst, E, Etot, H, as1, ad1, m1);
  k_edge_sum<<<((size_t)Etot * H + TB - 1) / TB, TB, 0, stream>>>(src, dst, E, Etot, H, as1, ad1, m1, s1);
  k_edge_scat4<<<Etot, TB, 0, stream>>>(src, dst, E, as1, ad1, m1, s1, h1, out1);

  // ---- bias + ELU -> bf16 layer-2 input ----
  k_post1<<<FILLB, TB, 0, stream>>>(out1, b1, x2b, (size_t)N * F1);

  // ---- GEMM2: h2 = x2 @ W2 ----
  {
    int mtiles = (N + 15) / 16;
    k_wmma_gemm_bf16<<<dim3((mtiles + 7) / 8, OUT / 64), TB, 0, stream>>>(x2b, w2b, h2, N, OUT, F1);
  }

  // ---- layer-2 attention (1 head) ----
  k_att<<<((size_t)N + 7) / 8, TB, 0, stream>>>(h2, at_s2, at_d2, as2, ad2, N, 1);

  // ---- layer-2 segment softmax + scatter into d_out ----
  k_fill_i32<<<FILLB, TB, 0, stream>>>(m2, (int)0x80000000, (size_t)N);
  k_fill_f32<<<FILLB, TB, 0, stream>>>(s2, 0.f, (size_t)N);
  k_fill_f32<<<FILLB, TB, 0, stream>>>(out, 0.f, (size_t)N * OUT);
  k_edge_max<<<((size_t)Etot + TB - 1) / TB, TB, 0, stream>>>(src, dst, E, Etot, 1, as2, ad2, m2);
  k_edge_sum<<<((size_t)Etot + TB - 1) / TB, TB, 0, stream>>>(src, dst, E, Etot, 1, as2, ad2, m2, s2);
  k_edge_scat1<<<(Etot + 3) / 4, TB, 0, stream>>>(src, dst, E, Etot, as2, ad2, m2, s2, h2, out);

  // ---- final bias ----
  k_post2<<<FILLB, TB, 0, stream>>>(out, b2, (size_t)N * OUT);
}
